// LinearAttentionBlock_12601434046606
// MI455X (gfx1250) — matrix operand order repaired and checked by reading, not compiled, not run on hardware
//
#include <hip/hip_runtime.h>
#include <hip/hip_bf16.h>

// ---------------- problem constants ----------------
constexpr int BATCH  = 32;
constexpr int CH     = 512;   // C = DIM
constexpr int NTOK   = 1024;  // H*W
constexpr int HID    = 512;   // HEADS * DIM_HEAD
constexpr int HEADS  = 8;
constexpr int DH     = 64;

// ---------------- WMMA types ----------------
typedef __attribute__((ext_vector_type(16))) __bf16 v16bf;
typedef __attribute__((ext_vector_type(8)))  float  v8f;

union FragAB {
  v16bf v;
  unsigned short u[16];
};

__device__ inline unsigned short f2bf(float x) {
  unsigned int u = __float_as_uint(x);
  u += 0x7FFFu + ((u >> 16) & 1u);   // round-to-nearest-even
  return (unsigned short)(u >> 16);
}
__device__ inline unsigned int pack2bf(float lo, float hi) {
  return (unsigned int)f2bf(lo) | ((unsigned int)f2bf(hi) << 16);
}
__device__ inline float bf2f(unsigned short h) {
  return __uint_as_float(((unsigned int)h) << 16);
}

__device__ inline v8f wmma_bf16(v16bf a, v16bf b, v8f c) {
  // (neg_a, A, neg_b, B, c_mod, C, reuse_a, reuse_b)
  return __builtin_amdgcn_wmma_f32_16x16x32_bf16(false, a, false, b, (short)0, c,
                                                 false, false);
}

// Low 32 bits of a flat shared pointer == LDS byte address (ISA 10.2).
__device__ inline unsigned int lds_lo32(const void* p) {
  return (unsigned int)(unsigned long long)p;
}

// Async copy one dword global->LDS (per-lane addresses), tracked by ASYNCcnt.
__device__ inline void async_b32(unsigned int lds_addr, const void* gptr) {
  unsigned long long ga = (unsigned long long)gptr;
  asm volatile("global_load_async_to_lds_b32 %0, %1, off"
               :
               : "v"(lds_addr), "v"(ga)
               : "memory");
}
__device__ inline void wait_asynccnt0() {
  asm volatile("s_wait_asynccnt 0x0" ::: "memory");
}

// A fragment: 16x32 bf16 tile, base at (m=0,k=0), row-major, ld in ushorts.
__device__ inline v16bf load_a_frag(const unsigned short* base, int ld, int lane) {
  const int m    = lane & 15;
  const int half = (lane >> 4) & 1;
  const unsigned short* p0 = base + m * ld + half * 8;
  const unsigned short* p1 = base + m * ld + 16 + half * 8;
  FragAB f;
#pragma unroll
  for (int i = 0; i < 8; ++i) {
    f.u[i]     = p0[i];
    f.u[8 + i] = p1[i];
  }
  return f.v;
}

// B fragment: 32x16 bf16 stored transposed in LDS as Bt[n][k] (k contiguous).
__device__ inline v16bf load_b_frag(const unsigned short* base, int ld, int lane) {
  const int n    = lane & 15;
  const int half = (lane >> 4) & 1;
  const unsigned short* p0 = base + n * ld + half * 8;
  const unsigned short* p1 = base + n * ld + 16 + half * 8;
  FragAB f;
#pragma unroll
  for (int i = 0; i < 8; ++i) {
    f.u[i]     = p0[i];
    f.u[8 + i] = p1[i];
  }
  return f.v;
}

constexpr int LDPW = 17;        // row stride in dwords (odd -> no bank conflicts)
constexpr int LDP  = 2 * LDPW;  // 34 ushorts

// ============================================================================
// Kernel 1: QKV projection.  xt = x[b]^T (transpose folded into LDS staging).
// Block: 128 tokens x 64 out-cols, 256 thr (8 waves). Q,K,V share the A tile.
// ============================================================================
__global__ __launch_bounds__(256) void qkv_kernel(
    const float* __restrict__ x, const float* __restrict__ Wq,
    const float* __restrict__ Wk, const float* __restrict__ Wv,
    unsigned short* __restrict__ Qb, unsigned short* __restrict__ Kb,
    unsigned short* __restrict__ Vb) {
  const int t    = threadIdx.x;
  const int lane = t & 31;
  const int wave = t >> 5;
  const int n0   = blockIdx.x * 128;
  const int c0   = blockIdx.y * 64;
  const int b    = blockIdx.z;

  __shared__ unsigned int As32[128 * LDPW];    // [m token][k chan pair]
  __shared__ unsigned int Bs32[3][64 * LDPW];  // [n outcol][k chan pair]
  const unsigned short* As = (const unsigned short*)As32;

  v8f accQ[4], accK[4], accV[4];
  const v8f vzero = {};
#pragma unroll
  for (int i = 0; i < 4; ++i) { accQ[i] = vzero; accK[i] = vzero; accV[i] = vzero; }

  const float* wsrc[3] = {Wq, Wk, Wv};

  for (int k0 = 0; k0 < CH; k0 += 32) {
    // stage A: x[b][k0+2kp..+1][n0+mm] -> As[mm][2kp] packed u32
#pragma unroll
    for (int i = 0; i < 8; ++i) {
      const int idx = t + 256 * i;
      const int mm  = idx & 127;
      const int kp  = idx >> 7;  // 0..15
      const size_t g = ((size_t)(b * CH + k0 + 2 * kp)) * NTOK + n0 + mm;
      As32[mm * LDPW + kp] = pack2bf(x[g], x[g + NTOK]);
      if (k0 + 32 < CH) __builtin_prefetch(&x[g + (size_t)32 * NTOK], 0, 1);
    }
    // stage B (x3): W[k0+2kp..+1][c0+nn] -> Bs[w][nn][2kp] packed u32
#pragma unroll
    for (int wsel = 0; wsel < 3; ++wsel) {
      const float* W = wsrc[wsel];
#pragma unroll
      for (int i = 0; i < 4; ++i) {
        const int idx = t + 256 * i;
        const int nn  = idx & 63;
        const int kp  = idx >> 6;
        const size_t g = (size_t)(k0 + 2 * kp) * HID + c0 + nn;
        Bs32[wsel][nn * LDPW + kp] = pack2bf(W[g], W[g + HID]);
      }
    }
    __syncthreads();

    const v16bf a = load_a_frag(&As[(wave * 16) * LDP], LDP, lane);
#pragma unroll
    for (int nt = 0; nt < 4; ++nt) {
      const v16bf bq = load_b_frag((const unsigned short*)&Bs32[0][(nt * 16) * LDPW], LDP, lane);
      accQ[nt] = wmma_bf16(a, bq, accQ[nt]);
      const v16bf bk = load_b_frag((const unsigned short*)&Bs32[1][(nt * 16) * LDPW], LDP, lane);
      accK[nt] = wmma_bf16(a, bk, accK[nt]);
      const v16bf bv = load_b_frag((const unsigned short*)&Bs32[2][(nt * 16) * LDPW], LDP, lane);
      accV[nt] = wmma_bf16(a, bv, accV[nt]);
    }
    __syncthreads();
  }

  const int half = lane >> 4;
  const int col  = lane & 15;
#pragma unroll
  for (int nt = 0; nt < 4; ++nt) {
#pragma unroll
    for (int r = 0; r < 8; ++r) {
      const int m    = wave * 16 + r + 8 * half;
      const size_t o = ((size_t)b * NTOK + n0 + m) * HID + c0 + nt * 16 + col;
      Qb[o] = f2bf(accQ[nt][r]);
      Kb[o] = f2bf(accK[nt][r]);
      Vb[o] = f2bf(accV[nt][r]);
    }
  }
}

// ============================================================================
// Kernel 2a: coalesced online-softmax partials over token chunks of 256.
// Thread t = column c (512 threads); grid (chunk=4, batch). All reads are
// row-contiguous across the block -> fully coalesced.
// ============================================================================
__global__ __launch_bounds__(512) void ksm_partial_kernel(
    const unsigned short* __restrict__ Kb, float* __restrict__ pmax,
    float* __restrict__ psum) {
  const int c  = threadIdx.x;       // column 0..511
  const int ch = blockIdx.x;        // token chunk 0..3
  const int b  = blockIdx.y;

  const unsigned short* base = Kb + ((size_t)b * NTOK + ch * 256) * HID + c;

  float m = -3.4e38f;
  float s = 0.0f;
  for (int n = 0; n < 256; ++n) {
    const float v = bf2f(base[(size_t)n * HID]);
    if (v > m) {
      s = s * __expf(m - v) + 1.0f;
      m = v;
    } else {
      s += __expf(v - m);
    }
  }
  const size_t o = ((size_t)ch * BATCH + b) * HID + c;
  pmax[o] = m;
  psum[o] = s;
}

// ============================================================================
// Kernel 2b: combine the 4 chunk partials per column, normalize K in place.
// ============================================================================
__global__ __launch_bounds__(512) void ksm_norm_kernel(
    unsigned short* __restrict__ Kb, const float* __restrict__ pmax,
    const float* __restrict__ psum) {
  const int c  = threadIdx.x;
  const int ch = blockIdx.x;
  const int b  = blockIdx.y;

  float M = -3.4e38f;
#pragma unroll
  for (int j = 0; j < 4; ++j)
    M = fmaxf(M, pmax[((size_t)j * BATCH + b) * HID + c]);
  float S = 0.0f;
#pragma unroll
  for (int j = 0; j < 4; ++j) {
    const size_t o = ((size_t)j * BATCH + b) * HID + c;
    S += psum[o] * __expf(pmax[o] - M);
  }
  const float inv = 1.0f / S;

  unsigned short* base = Kb + ((size_t)b * NTOK + ch * 256) * HID + c;
  for (int n = 0; n < 256; ++n) {
    const float v = bf2f(base[(size_t)n * HID]);
    base[(size_t)n * HID] = f2bf(__expf(v - M) * inv);
  }
}

// ============================================================================
// Kernel 3: context[b,h] = Ks[b,:,h]^T (64x1024) @ V[b,:,h] (1024x64) -> 64x64.
// One block per (h,b); 16 output tiles over 8 waves (2 each); K-loop of 32.
// ============================================================================
__global__ __launch_bounds__(256) void context_kernel(
    const unsigned short* __restrict__ Ks, const unsigned short* __restrict__ Vb,
    unsigned short* __restrict__ Cx) {
  const int t    = threadIdx.x;
  const int lane = t & 31;
  const int wave = t >> 5;
  const int h    = blockIdx.x;
  const int b    = blockIdx.y;

  __shared__ unsigned int As32[64 * LDPW];  // [d][tok pair]  (A = Ks^T)
  __shared__ unsigned int Bt32[64 * LDPW];  // [v][tok pair]

  v8f acc0 = {}, acc1 = {};
  const int t0 = wave * 2, t1 = wave * 2 + 1;
  const int mt0 = t0 >> 2, nt0 = t0 & 3;
  const int mt1 = t1 >> 2, nt1 = t1 & 3;

  const size_t rowbase = (size_t)b * NTOK * HID + h * DH;

  for (int n0 = 0; n0 < NTOK; n0 += 32) {
#pragma unroll
    for (int i = 0; i < 4; ++i) {
      const int idx = t + 256 * i;
      const int dd  = idx & 63;  // head-dim element
      const int kp  = idx >> 6;  // token pair within step
      const size_t g = rowbase + (size_t)(n0 + 2 * kp) * HID + dd;
      As32[dd * LDPW + kp] =
          (unsigned int)Ks[g] | ((unsigned int)Ks[g + HID] << 16);
      Bt32[dd * LDPW + kp] =
          (unsigned int)Vb[g] | ((unsigned int)Vb[g + HID] << 16);
    }
    __syncthreads();
    const v16bf a0 = load_a_frag((const unsigned short*)&As32[(mt0 * 16) * LDPW], LDP, lane);
    const v16bf b0 = load_b_frag((const unsigned short*)&Bt32[(nt0 * 16) * LDPW], LDP, lane);
    acc0 = wmma_bf16(a0, b0, acc0);
    const v16bf a1 = load_a_frag((const unsigned short*)&As32[(mt1 * 16) * LDPW], LDP, lane);
    const v16bf b1 = load_b_frag((const unsigned short*)&Bt32[(nt1 * 16) * LDPW], LDP, lane);
    acc1 = wmma_bf16(a1, b1, acc1);
    __syncthreads();
  }

  const int half = lane >> 4;
  const int col  = lane & 15;
  const size_t cbase = ((size_t)(b * HEADS + h)) * DH * DH;
#pragma unroll
  for (int r = 0; r < 8; ++r) {
    Cx[cbase + (size_t)(mt0 * 16 + r + 8 * half) * DH + nt0 * 16 + col] = f2bf(acc0[r]);
    Cx[cbase + (size_t)(mt1 * 16 + r + 8 * half) * DH + nt1 * 16 + col] = f2bf(acc1[r]);
  }
}

// ============================================================================
// Kernel 4: attn_out[b,:,h] = Q[b,:,h] (1024x64) @ context[b,h] (64x64).
// A tile staged via async global->LDS dword copies (ASYNCcnt).
// ============================================================================
__global__ __launch_bounds__(256) void attnout_kernel(
    const unsigned short* __restrict__ Qb, const unsigned short* __restrict__ Cx,
    unsigned short* __restrict__ AO) {
  const int t    = threadIdx.x;
  const int lane = t & 31;
  const int wave = t >> 5;
  const int n0   = blockIdx.x * 128;
  const int h    = blockIdx.y;
  const int b    = blockIdx.z;

  __shared__ unsigned int As32[128 * LDPW];  // [m tok][k d pair]
  __shared__ unsigned int Bt32[64 * LDPW];   // [v][k d pair]

  const v8f vzero = {};
  v8f acc[4];
#pragma unroll
  for (int i = 0; i < 4; ++i) acc[i] = vzero;

  const size_t cbase = ((size_t)(b * HEADS + h)) * DH * DH;
  const unsigned int* Q32 = (const unsigned int*)Qb;

  for (int k0 = 0; k0 < DH; k0 += 32) {
    // A: Q[b][n0+mm][h*64 + k0 + 2dp..+1] -> As[mm][dp]  (async dword copies)
#pragma unroll
    for (int i = 0; i < 8; ++i) {
      const int idx = t + 256 * i;
      const int mm  = idx >> 4;
      const int dp  = idx & 15;
      const size_t g32 =
          ((size_t)b * NTOK + n0 + mm) * (HID / 2) + (h * DH + k0) / 2 + dp;
      async_b32(lds_lo32(&As32[mm * LDPW + dp]), &Q32[g32]);
    }
    // B: Cx[b,h][k0+2dp..+1][vv] -> Bt[vv][dp]
#pragma unroll
    for (int i = 0; i < 4; ++i) {
      const int idx = t + 256 * i;
      const int vv  = idx & 63;
      const int dp  = idx >> 6;
      const size_t ci = cbase + (size_t)(k0 + 2 * dp) * DH + vv;
      Bt32[vv * LDPW + dp] =
          (unsigned int)Cx[ci] | ((unsigned int)Cx[ci + DH] << 16);
    }
    wait_asynccnt0();
    __syncthreads();
    const v16bf a = load_a_frag((const unsigned short*)&As32[(wave * 16) * LDPW], LDP, lane);
#pragma unroll
    for (int nt = 0; nt < 4; ++nt) {
      const v16bf bb = load_b_frag((const unsigned short*)&Bt32[(nt * 16) * LDPW], LDP, lane);
      acc[nt] = wmma_bf16(a, bb, acc[nt]);
    }
    __syncthreads();
  }

  const int half = lane >> 4;
  const int col  = lane & 15;
#pragma unroll
  for (int nt = 0; nt < 4; ++nt) {
#pragma unroll
    for (int r = 0; r < 8; ++r) {
      const int m = wave * 16 + r + 8 * half;
      AO[((size_t)b * NTOK + n0 + m) * HID + h * DH + nt * 16 + col] =
          f2bf(acc[nt][r]);
    }
  }
}

// ============================================================================
// Kernel 5: out[b,c,n] = AO[b,n,:] @ Wp[:,c] + bp[c]  (transposed f32 store).
// A tile staged via async global->LDS dword copies.
// ============================================================================
__global__ __launch_bounds__(256) void proj_kernel(
    const unsigned short* __restrict__ AO, const float* __restrict__ Wp,
    const float* __restrict__ bp, float* __restrict__ out) {
  const int t    = threadIdx.x;
  const int lane = t & 31;
  const int wave = t >> 5;
  const int n0   = blockIdx.x * 128;
  const int c0   = blockIdx.y * 64;
  const int b    = blockIdx.z;

  __shared__ unsigned int As32[128 * LDPW];
  __shared__ unsigned int Bt32[64 * LDPW];

  const v8f vzero = {};
  v8f acc[4];
#pragma unroll
  for (int i = 0; i < 4; ++i) acc[i] = vzero;

  const unsigned int* AO32 = (const unsigned int*)AO;

  for (int k0 = 0; k0 < HID; k0 += 32) {
    // A: AO[b][n0+mm][k0 + 2kp..+1] -> As[mm][kp]  (async dword copies)
#pragma unroll
    for (int i = 0; i < 8; ++i) {
      const int idx = t + 256 * i;
      const int mm  = idx >> 4;
      const int kp  = idx & 15;
      const size_t g32 = ((size_t)b * NTOK + n0 + mm) * (HID / 2) + k0 / 2 + kp;
      async_b32(lds_lo32(&As32[mm * LDPW + kp]), &AO32[g32]);
    }
    // B: Wp[k0+2kp..+1][c0+nn] -> Bt[nn][kp]
#pragma unroll
    for (int i = 0; i < 4; ++i) {
      const int idx = t + 256 * i;
      const int nn  = idx & 63;
      const int kp  = idx >> 6;
      const size_t g = (size_t)(k0 + 2 * kp) * CH + c0 + nn;
      Bt32[nn * LDPW + kp] = pack2bf(Wp[g], Wp[g + CH]);
    }
    wait_asynccnt0();
    __syncthreads();
    const v16bf a = load_a_frag((const unsigned short*)&As32[(wave * 16) * LDPW], LDP, lane);
#pragma unroll
    for (int nt = 0; nt < 4; ++nt) {
      const v16bf bb = load_b_frag((const unsigned short*)&Bt32[(nt * 16) * LDPW], LDP, lane);
      acc[nt] = wmma_bf16(a, bb, acc[nt]);
    }
    __syncthreads();
  }

  const int half = lane >> 4;
  const int col  = lane & 15;
#pragma unroll
  for (int nt = 0; nt < 4; ++nt) {
    const int cc = c0 + nt * 16 + col;
    const float bias = bp[cc];
#pragma unroll
    for (int r = 0; r < 8; ++r) {
      const int m = wave * 16 + r + 8 * half;
      out[((size_t)b * CH + cc) * NTOK + n0 + m] = acc[nt][r] + bias;
    }
  }
}

// ============================================================================
extern "C" void kernel_launch(void* const* d_in, const int* in_sizes, int n_in,
                              void* d_out, int out_size, void* d_ws, size_t ws_size,
                              hipStream_t stream) {
  const float* x  = (const float*)d_in[0];
  const float* Wq = (const float*)d_in[1];
  const float* Wk = (const float*)d_in[2];
  const float* Wv = (const float*)d_in[3];
  const float* Wp = (const float*)d_in[4];
  const float* bp = (const float*)d_in[5];
  float* out = (float*)d_out;

  // bf16 workspace layout (~130.5 MB total)
  const size_t elems = (size_t)BATCH * NTOK * HID;  // 16M elems each
  unsigned short* Qb = (unsigned short*)d_ws;
  unsigned short* Kb = Qb + elems;
  unsigned short* Vb = Kb + elems;
  unsigned short* Cx = Vb + elems;                  // B*H*64*64 = 1M elems
  unsigned short* AO = Cx + (size_t)BATCH * HEADS * DH * DH;
  float* pmax = (float*)(AO + elems);               // 4*B*HID f32 partial max
  float* psum = pmax + (size_t)4 * BATCH * HID;     // 4*B*HID f32 partial sum

  qkv_kernel<<<dim3(NTOK / 128, HID / 64, BATCH), 256, 0, stream>>>(
      x, Wq, Wk, Wv, Qb, Kb, Vb);
  ksm_partial_kernel<<<dim3(NTOK / 256, BATCH), 512, 0, stream>>>(Kb, pmax, psum);
  ksm_norm_kernel<<<dim3(NTOK / 256, BATCH), 512, 0, stream>>>(Kb, pmax, psum);
  context_kernel<<<dim3(HEADS, BATCH), 256, 0, stream>>>(Kb, Vb, Cx);
  attnout_kernel<<<dim3(NTOK / 128, HEADS, BATCH), 256, 0, stream>>>(Qb, Cx, AO);
  proj_kernel<<<dim3(NTOK / 128, CH / 64, BATCH), 256, 0, stream>>>(AO, Wp, bp, out);
}
